// CSFN_75737453298030
// MI455X (gfx1250) — matrix operand order repaired and checked by reading, not compiled, not run on hardware
//
#include <hip/hip_runtime.h>
#include <hip/hip_bf16.h>

#define DEPTHL 12
#define HEADS 12
#define DIMC 768
#define HDIM 64
#define NTOK 199
#define BATCH 32
#define NPAD 224               // padded key dim of S,P and of vT
#define RPAD 208               // padded query/row dim of S,P
#define SEQ_ALL (BATCH*NTOK)   // 6368 (multiple of 16)

#define BMT 128
#define BNT 128
#define LDK 40                 // LDS row stride (f16 elems): 80B, conflict-free & 16B aligned

typedef __attribute__((ext_vector_type(16))) _Float16 v16h;
typedef __attribute__((ext_vector_type(8)))  float    v8f;
typedef int i32x4 __attribute__((vector_size(16)));

union Frag { uint4 u[2]; v16h v; };

// ---- CDNA5 async global->LDS (guarded; falls back to sync copy) ----
#if defined(__has_builtin)
#  if __has_builtin(__builtin_amdgcn_global_load_async_to_lds_b128) && \
      __has_builtin(__builtin_amdgcn_s_wait_asynccnt)
#    define CDNA5_ASYNC 1
#  endif
#endif
#ifndef CDNA5_ASYNC
#  define CDNA5_ASYNC 0
#endif

static __device__ __forceinline__ void copy16(void* lds, const void* g) {
#if CDNA5_ASYNC
  __builtin_amdgcn_global_load_async_to_lds_b128(
      (__attribute__((address_space(1))) i32x4*)(g),
      (__attribute__((address_space(3))) i32x4*)(lds), 0, 0);
#else
  *(uint4*)lds = *(const uint4*)g;
#endif
}

// ---------------- reductions (wave32) ----------------
static __device__ __forceinline__ float blockReduceSum(float v, float* sh) {
  #pragma unroll
  for (int o = 16; o > 0; o >>= 1) v += __shfl_down(v, o, 32);
  int lane = threadIdx.x & 31, w = threadIdx.x >> 5;
  if (lane == 0) sh[w] = v;
  __syncthreads();
  float r = (threadIdx.x < 8) ? sh[threadIdx.x] : 0.f;
  if (w == 0) {
    #pragma unroll
    for (int o = 4; o > 0; o >>= 1) r += __shfl_down(r, o, 32);
    if (lane == 0) sh[0] = r;
  }
  __syncthreads();
  float out = sh[0];
  __syncthreads();
  return out;
}

static __device__ __forceinline__ float blockReduceMax(float v, float* sh) {
  #pragma unroll
  for (int o = 16; o > 0; o >>= 1) v = fmaxf(v, __shfl_down(v, o, 32));
  int lane = threadIdx.x & 31, w = threadIdx.x >> 5;
  if (lane == 0) sh[w] = v;
  __syncthreads();
  float r = (threadIdx.x < 8) ? sh[threadIdx.x] : -3.0e38f;
  if (w == 0) {
    #pragma unroll
    for (int o = 4; o > 0; o >>= 1) r = fmaxf(r, __shfl_down(r, o, 32));
    if (lane == 0) sh[0] = r;
  }
  __syncthreads();
  float out = sh[0];
  __syncthreads();
  return out;
}

// ---------------- batched WMMA GEMM ----------------
// C[z][m][n] = alpha * sum_k A[z][m][k]*Bt[z][n][k] (+bias[n]) (+resid[m][n]) (gelu?)
// A:  f16 [M][K] row-major (lda); per-z offset = (z/H)*sAz + (z%H)*sAh   (elements)
// Bt: f16 stored TRANSPOSED [N][K] row-major (ldb); same per-z scheme.
// Unguarded loads: callers guarantee over-reads stay inside the workspace
// (slack region at the end); garbage rows/cols only affect masked outputs.
// Macro tile 128x128x32, 8 waves; wave tile 32x64 (2x4 WMMA tiles).
// Double-buffered LDS with async global->LDS prefetch of the next K-step.
__global__ __launch_bounds__(256) void wmma_gemm_k(
    const _Float16* __restrict__ A, long lda, long sAz, long sAh,
    const _Float16* __restrict__ Bt, long ldb, long sBz, long sBh,
    const float* __restrict__ bias,
    const float* __restrict__ resid, long ldr,
    float* __restrict__ Cf, _Float16* __restrict__ Cb,
    long ldc, long sCz, long sCh,
    int M, int N, int K, int H,
    float alpha, int fuseGelu)
{
  __shared__ __align__(16) _Float16 As[2][BMT * LDK];
  __shared__ __align__(16) _Float16 Bs[2][BNT * LDK];

  const int z = blockIdx.z;
  const int bb = z / H, hh = z % H;
  const _Float16* Ab = A  + (long)bb * sAz + (long)hh * sAh;
  const _Float16* Bb = Bt + (long)bb * sBz + (long)hh * sBh;
  const long cOff = (long)bb * sCz + (long)hh * sCh;

  const int m0 = blockIdx.x * BMT;
  const int n0 = blockIdx.y * BNT;
  const int t = threadIdx.x;
  const int wave = t >> 5, lane = t & 31;
  const int wm = wave >> 1, wn = wave & 1;   // 4 x 2 waves
  const int l16 = lane & 15;
  const bool hi = lane >= 16;

  // staging chunk geometry: 512 16B-chunks per 128x32 tile; 2 chunks/thread/tile
  const int c0row = t >> 2, c0seg = (t & 3) << 3;
  const int c1row = (t + 256) >> 2, c1seg = ((t + 256) & 3) << 3;

  auto stage = [&](int buf, int k0) {
    copy16(&As[buf][c0row * LDK + c0seg], Ab + (long)(m0 + c0row) * lda + (k0 + c0seg));
    copy16(&As[buf][c1row * LDK + c1seg], Ab + (long)(m0 + c1row) * lda + (k0 + c1seg));
    copy16(&Bs[buf][c0row * LDK + c0seg], Bb + (long)(n0 + c0row) * ldb + (k0 + c0seg));
    copy16(&Bs[buf][c1row * LDK + c1seg], Bb + (long)(n0 + c1row) * ldb + (k0 + c1seg));
  };

  v8f zero8 = {0.f,0.f,0.f,0.f,0.f,0.f,0.f,0.f};
  v8f acc[2][4];
  #pragma unroll
  for (int mt = 0; mt < 2; ++mt)
    #pragma unroll
    for (int nt = 0; nt < 4; ++nt) acc[mt][nt] = zero8;

  stage(0, 0);
  int pb = 0;
  for (int k0 = 0; k0 < K; k0 += 32) {
    const bool hasNext = (k0 + 32) < K;
    if (hasNext) stage(pb ^ 1, k0 + 32);
#if CDNA5_ASYNC
    if (hasNext) __builtin_amdgcn_s_wait_asynccnt(4);  // let the 4 just-issued fly
    else         __builtin_amdgcn_s_wait_asynccnt(0);
#endif
    __syncthreads();   // all waves' current-buffer DMA complete & visible

    Frag af[2], bfr[4];
    #pragma unroll
    for (int mt = 0; mt < 2; ++mt) {
      int row = wm * 32 + mt * 16 + l16;
      int kg = hi ? 8 : 0;   // A layout: lanes<16 -> K 0..7 & 16..23; lanes>=16 -> +8
      af[mt].u[0] = *(const uint4*)(&As[pb][row * LDK + kg]);
      af[mt].u[1] = *(const uint4*)(&As[pb][row * LDK + kg + 16]);
    }
    #pragma unroll
    for (int nt = 0; nt < 4; ++nt) {
      int col = wn * 64 + nt * 16 + l16;
      int kb = hi ? 8 : 0;   // B layout: lanes<16 -> K 0..7 & 16..23; lanes>=16 -> +8
      bfr[nt].u[0] = *(const uint4*)(&Bs[pb][col * LDK + kb]);
      bfr[nt].u[1] = *(const uint4*)(&Bs[pb][col * LDK + kb + 16]);
    }
    #pragma unroll
    for (int mt = 0; mt < 2; ++mt)
      #pragma unroll
      for (int nt = 0; nt < 4; ++nt)
        acc[mt][nt] = __builtin_amdgcn_wmma_f32_16x16x32_f16(
            false, af[mt].v, false, bfr[nt].v, (short)0, acc[mt][nt], false, false);

    __syncthreads();   // all waves done reading before this buffer is overwritten
    pb ^= 1;
  }

  // store: C layout - VGPR r: M = r (+8 for lanes>=16), N = lane&15
  #pragma unroll
  for (int mt = 0; mt < 2; ++mt) {
    #pragma unroll
    for (int nt = 0; nt < 4; ++nt) {
      int rowbase = m0 + wm * 32 + mt * 16 + (hi ? 8 : 0);
      int col = n0 + wn * 64 + nt * 16 + l16;
      if (col >= N) continue;
      float bv = bias ? bias[col] : 0.f;
      #pragma unroll
      for (int r = 0; r < 8; ++r) {
        int m = rowbase + r;
        if (m >= M) continue;
        float v = alpha * acc[mt][nt][r] + bv;
        if (resid) v += resid[(long)m * ldr + col];
        if (fuseGelu) v = 0.5f * v * (1.f + erff(v * 0.70710678118654752f));
        long idx = cOff + (long)m * ldc + col;
        if (Cf) Cf[idx] = v;
        if (Cb) Cb[idx] = (_Float16)v;
      }
    }
  }
}

// ---------------- layernorm (row of 768) ----------------
__global__ __launch_bounds__(256) void ln_k(
    const float* __restrict__ x, const float* __restrict__ g, const float* __restrict__ bta,
    _Float16* __restrict__ outH, float* __restrict__ outF)
{
  __shared__ float sh[8];
  long row = blockIdx.x;
  const float* xr = x + row * DIMC;
  int t = threadIdx.x;
  float vals[3];
  float s = 0.f, s2 = 0.f;
  #pragma unroll
  for (int i = 0; i < 3; ++i) {
    float v = xr[t + i * 256];
    vals[i] = v; s += v; s2 += v * v;
  }
  float tot  = blockReduceSum(s, sh);
  float tot2 = blockReduceSum(s2, sh);
  float mean = tot * (1.f / DIMC);
  float var  = tot2 * (1.f / DIMC) - mean * mean;
  float inv  = rsqrtf(var + 1e-5f);
  #pragma unroll
  for (int i = 0; i < 3; ++i) {
    int col = t + i * 256;
    float v = (vals[i] - mean) * inv * g[col] + bta[col];
    if (outH) outH[row * DIMC + col] = (_Float16)v;
    if (outF) outF[row * DIMC + col] = v;
  }
}

// ---------------- convert f32 -> f16 (no transpose) ----------------
__global__ void cvt_k(const float* __restrict__ in, _Float16* __restrict__ out, long n) {
  long i = (long)blockIdx.x * 256 + threadIdx.x;
  if (i < n) out[i] = (_Float16)in[i];
}

// transpose-convert: in f32 [L][K][N] -> out f16 [L][N][K]
__global__ void wtrans_k(const float* __restrict__ in, _Float16* __restrict__ out,
                         int Kd, int Nd, long total) {
  long i = (long)blockIdx.x * 256 + threadIdx.x;
  if (i >= total) return;
  long per = (long)Kd * Nd;
  long l = i / per, r = i % per;
  long n = r / Kd, k = r % Kd;
  out[i] = (_Float16)in[l * per + k * (long)Nd + n];
}

// transpose V slices: vT[b][h][d][m] = qkv[b*199+m][1536 + h*64 + d], zero pad m>=199
__global__ void vtrans_k(const _Float16* __restrict__ qkv, _Float16* __restrict__ vT) {
  long i = (long)blockIdx.x * 256 + threadIdx.x;
  const long TOT = (long)BATCH * HEADS * HDIM * NPAD;
  if (i >= TOT) return;
  int m = (int)(i % NPAD);
  long r = i / NPAD;
  int d = (int)(r % HDIM);
  long bh = r / HDIM;
  int h = (int)(bh % HEADS);
  int b = (int)(bh / HEADS);
  _Float16 v = (_Float16)0.f;
  if (m < NTOK) v = qkv[((long)b * NTOK + m) * 2304 + 1536 + h * HDIM + d];
  vT[i] = v;
}

// im2col: A[b*196+t][c*256+p*16+q] = x[b][c][h*16+p][w*16+q]
__global__ void im2col_k(const float* __restrict__ x, _Float16* __restrict__ A) {
  long i = (long)blockIdx.x * 256 + threadIdx.x;
  const long TOT = (long)BATCH * 196 * 768;
  if (i >= TOT) return;
  int k = (int)(i % 768);
  long mt = i / 768;
  int b = (int)(mt / 196), tt = (int)(mt % 196);
  int c = k >> 8, p = (k >> 4) & 15, q = k & 15;
  int hh = tt / 14, ww = tt % 14;
  int r = hh * 16 + p, cc = ww * 16 + q;
  A[i] = (_Float16)x[(((long)b * 3 + c) * 224 + r) * 224 + cc];
}

// assemble x = concat(cls, tok, loc, loc_aug) + concat(pos, loc_e, laug_e)
__global__ __launch_bounds__(256) void assemble_k(
    const float* __restrict__ tok, const float* __restrict__ cls,
    const float* __restrict__ loc, const float* __restrict__ laug,
    const float* __restrict__ pos, const float* __restrict__ lpos,
    const float* __restrict__ lapos, float* __restrict__ x)
{
  int row = blockIdx.x;
  int b = row / NTOK, n = row % NTOK;
  #pragma unroll
  for (int i = 0; i < 3; ++i) {
    int col = threadIdx.x + i * 256;
    float v, pv;
    if (n == 0)        { v = cls[col];  pv = pos[col]; }
    else if (n <= 196) { v = tok[((long)b * 196 + (n - 1)) * 768 + col]; pv = pos[(long)n * 768 + col]; }
    else if (n == 197) { v = loc[col];  pv = lpos[col]; }
    else               { v = laug[col]; pv = lapos[col]; }
    x[(long)row * 768 + col] = v + pv;
  }
}

// mask_fuse[b][m] = max(sigmoid(mean_h S[b,h,197,m]), sigmoid(mean_h S[b,h,198,m]))
__global__ void mask_k(const float* __restrict__ S, float* __restrict__ mf) {
  int b = blockIdx.x;
  int t = threadIdx.x;
  if (t >= NPAD) return;
  float o = 0.f;
  if (t < NTOK) {
    float a1 = 0.f, a2 = 0.f;
    for (int h = 0; h < HEADS; ++h) {
      const float* base = S + ((long)(b * HEADS + h) * RPAD) * NPAD;
      a1 += base[(long)197 * NPAD + t];
      a2 += base[(long)198 * NPAD + t];
    }
    float m1 = 1.f / (1.f + expf(-a1 * (1.f / HEADS)));
    float m2 = 1.f / (1.f + expf(-a2 * (1.f / HEADS)));
    o = fmaxf(m1, m2);
  }
  mf[(long)b * NPAD + t] = o;
}

// softmax over key dim (199), optional post-softmax mask, write f16 probs (zero padded to 224)
__global__ __launch_bounds__(256) void softmax_k(
    const float* __restrict__ S, _Float16* __restrict__ P,
    const float* __restrict__ mf, int useMask)
{
  __shared__ float sh[8];
  int n = blockIdx.x;            // query row < NTOK
  int zh = blockIdx.y;           // b*HEADS + h
  int b = zh / HEADS;
  const float* row = S + ((long)zh * RPAD + n) * NPAD;
  _Float16* prow = P + ((long)zh * RPAD + n) * NPAD;
  int t = threadIdx.x;
  float v = (t < NTOK) ? row[t] : -3.0e38f;
  float mx = blockReduceMax(v, sh);
  float e = (t < NTOK) ? expf(v - mx) : 0.f;
  float sum = blockReduceSum(e, sh);
  float p = e / sum;
  if (useMask && t < NTOK) p *= mf[(long)b * NPAD + t];
  if (t < NPAD) prow[t] = (t < NTOK) ? (_Float16)p : (_Float16)0.f;
}

// ---------------- host side ----------------
static void gemm(hipStream_t s,
                 const _Float16* A, long lda, long sAz, long sAh,
                 const _Float16* Bt, long ldb, long sBz, long sBh,
                 const float* bias, const float* resid, long ldr,
                 float* Cf, _Float16* Cb, long ldc, long sCz, long sCh,
                 int M, int N, int K, int Z, int H,
                 float alpha, int gelu)
{
  dim3 g((M + BMT - 1) / BMT, (N + BNT - 1) / BNT, Z);
  wmma_gemm_k<<<g, 256, 0, s>>>(A, lda, sAz, sAh, Bt, ldb, sBz, sBh,
                                bias, resid, ldr, Cf, Cb, ldc, sCz, sCh,
                                M, N, K, H, alpha, gelu);
}

extern "C" void kernel_launch(void* const* d_in, const int* in_sizes, int n_in,
                              void* d_out, int out_size, void* d_ws, size_t ws_size,
                              hipStream_t stream)
{
  (void)in_sizes; (void)n_in; (void)out_size; (void)ws_size;
  const float* in_x     = (const float*)d_in[0];
  const float* patch_w  = (const float*)d_in[1];
  const float* patch_b  = (const float*)d_in[2];
  const float* cls_tok  = (const float*)d_in[3];
  const float* loc_tok  = (const float*)d_in[4];
  const float* loc_aug  = (const float*)d_in[5];
  const float* pos_emb  = (const float*)d_in[6];
  const float* loc_emb  = (const float*)d_in[7];
  const float* laug_emb = (const float*)d_in[8];
  const float* ln1_g = (const float*)d_in[9];
  const float* ln1_b = (const float*)d_in[10];
  const float* qkv_w = (const float*)d_in[11];
  const float* proj_w = (const float*)d_in[12];
  const float* proj_b = (const float*)d_in[13];
  const float* ln2_g = (const float*)d_in[14];
  const float* ln2_b = (const float*)d_in[15];
  const float* fc1_w = (const float*)d_in[16];
  const float* fc1_b = (const float*)d_in[17];
  const float* fc2_w = (const float*)d_in[18];
  const float* fc2_b = (const float*)d_in[19];
  const float* norm_g = (const float*)d_in[20];
  const float* norm_b = (const float*)d_in[21];

  char* base = (char*)d_ws;
  size_t off = 0;
  auto alloc = [&](size_t bytes) -> void* {
    void* p = base + off;
    off += (bytes + 255) & ~(size_t)255;
    return p;
  };

  const long n_wqkv = (long)DEPTHL * 768 * 2304;
  const long n_wproj = (long)DEPTHL * 768 * 768;
  const long n_wfc1 = (long)DEPTHL * 768 * 3072;
  const long n_wfc2 = (long)DEPTHL * 3072 * 768;
  const long n_wpt = (long)768 * 768;
  const long n_apatch = (long)BATCH * 196 * 768;
  const long n_x = (long)SEQ_ALL * 768;
  const long n_qkv = (long)SEQ_ALL * 2304;
  const long n_S = (long)BATCH * HEADS * RPAD * NPAD;
  const long n_vT = (long)BATCH * HEADS * HDIM * NPAD;
  const long n_mlp = (long)SEQ_ALL * 3072;

  // order matters: every over-readable buffer is followed by mapped memory,
  // and a slack region terminates the layout.
  _Float16* wqkvT  = (_Float16*)alloc(n_wqkv * 2);   // [12][2304][768]
  _Float16* wprojT = (_Float16*)alloc(n_wproj * 2);  // [12][768][768]
  _Float16* wfc1T  = (_Float16*)alloc(n_wfc1 * 2);   // [12][3072][768]
  _Float16* wfc2T  = (_Float16*)alloc(n_wfc2 * 2);   // [12][768][3072]
  _Float16* wpt    = (_Float16*)alloc(n_wpt * 2);    // [768][768] ([N][K] natively)
  _Float16* apatch = (_Float16*)alloc(n_apatch * 2);
  float*    tok    = (float*)alloc(n_apatch * 4);
  float*    xbuf   = (float*)alloc(n_x * 4);
  _Float16* hbuf   = (_Float16*)alloc(n_x * 2);
  _Float16* qkvb   = (_Float16*)alloc(n_qkv * 2);
  float*    Sbuf   = (float*)alloc(n_S * 4);
  _Float16* Pbuf   = (_Float16*)alloc(n_S * 2);
  _Float16* obuf   = (_Float16*)alloc(n_x * 2);
  _Float16* vT     = (_Float16*)alloc(n_vT * 2);
  _Float16* mlp    = (_Float16*)alloc(n_mlp * 2);
  float*    maskf  = (float*)alloc((long)BATCH * NPAD * 4);
  (void)alloc(2u << 20);  // slack: absorbs all unguarded tile over-reads

  // weight conversion (+transpose to [N][K])
  wtrans_k<<<(unsigned)((n_wqkv + 255) / 256), 256, 0, stream>>>(qkv_w, wqkvT, 768, 2304, n_wqkv);
  wtrans_k<<<(unsigned)((n_wproj + 255) / 256), 256, 0, stream>>>(proj_w, wprojT, 768, 768, n_wproj);
  wtrans_k<<<(unsigned)((n_wfc1 + 255) / 256), 256, 0, stream>>>(fc1_w, wfc1T, 768, 3072, n_wfc1);
  wtrans_k<<<(unsigned)((n_wfc2 + 255) / 256), 256, 0, stream>>>(fc2_w, wfc2T, 3072, 768, n_wfc2);
  cvt_k<<<(unsigned)((n_wpt + 255) / 256), 256, 0, stream>>>(patch_w, wpt, n_wpt);

  // patch embed
  im2col_k<<<(unsigned)((n_apatch + 255) / 256), 256, 0, stream>>>(in_x, apatch);
  gemm(stream, apatch, 768, 0, 0, wpt, 768, 0, 0,
       patch_b, nullptr, 0, tok, nullptr, 768, 0, 0,
       BATCH * 196, 768, 768, 1, 1, 1.f, 0);
  assemble_k<<<SEQ_ALL, 256, 0, stream>>>(tok, cls_tok, loc_tok, loc_aug,
                                          pos_emb, loc_emb, laug_emb, xbuf);

  const long sTok = (long)NTOK * 2304;      // per-batch stride in qkv buffer
  const long sSz = (long)HEADS * RPAD * NPAD;
  const long sSh = (long)RPAD * NPAD;
  const long sVz = (long)HEADS * HDIM * NPAD;
  const long sVh = (long)HDIM * NPAD;

  for (int i = 0; i < DEPTHL; ++i) {
    // LN1 -> h (f16)
    ln_k<<<SEQ_ALL, 256, 0, stream>>>(xbuf, ln1_g + (long)i * 768, ln1_b + (long)i * 768, hbuf, nullptr);
    // qkv = h @ Wqkv  -> f16 [6368][2304]
    gemm(stream, hbuf, 768, 0, 0, wqkvT + (long)i * 2304 * 768, 768, 0, 0,
         nullptr, nullptr, 0, nullptr, qkvb, 2304, 0, 0,
         SEQ_ALL, 2304, 768, 1, 1, 1.f, 0);
    // V slices -> transposed zero-padded [b,h][64][224]
    vtrans_k<<<(unsigned)((n_vT + 255) / 256), 256, 0, stream>>>(qkvb, vT);
    // scores S = scale * q k^T  (K-matrix rows are naturally [N][K])
    gemm(stream, qkvb, 2304, sTok, HDIM,
         qkvb + 768, 2304, sTok, HDIM,
         nullptr, nullptr, 0,
         Sbuf, nullptr, NPAD, sSz, sSh,
         NTOK, NTOK, HDIM, BATCH * HEADS, HEADS, 0.125f, 0);
    if (i >= 9) mask_k<<<BATCH, 256, 0, stream>>>(Sbuf, maskf);
    softmax_k<<<dim3(NTOK, BATCH * HEADS), 256, 0, stream>>>(Sbuf, Pbuf, maskf, (i >= 9) ? 1 : 0);
    // O = P @ V  -> f16 [6368][768] (heads re-interleaved)
    gemm(stream, Pbuf, NPAD, sSz, sSh,
         vT, NPAD, sVz, sVh,
         nullptr, nullptr, 0,
         nullptr, obuf, 768, (long)NTOK * 768, HDIM,
         NTOK, HDIM, NPAD, BATCH * HEADS, HEADS, 1.f, 0);
    // x = x + O @ Wproj + b
    gemm(stream, obuf, 768, 0, 0, wprojT + (long)i * 768 * 768, 768, 0, 0,
         proj_b + (long)i * 768, xbuf, 768, xbuf, nullptr, 768, 0, 0,
         SEQ_ALL, 768, 768, 1, 1, 1.f, 0);
    // LN2 -> h
    ln_k<<<SEQ_ALL, 256, 0, stream>>>(xbuf, ln2_g + (long)i * 768, ln2_b + (long)i * 768, hbuf, nullptr);
    // mlp = gelu(h @ W1 + b1) (f16)
    gemm(stream, hbuf, 768, 0, 0, wfc1T + (long)i * 3072 * 768, 768, 0, 0,
         fc1_b + (long)i * 3072, nullptr, 0, nullptr, mlp, 3072, 0, 0,
         SEQ_ALL, 3072, 768, 1, 1, 1.f, 1);
    // x = x + mlp @ W2 + b2
    gemm(stream, mlp, 3072, 0, 0, wfc2T + (long)i * 768 * 3072, 3072, 0, 0,
         fc2_b + (long)i * 768, xbuf, 768, xbuf, nullptr, 768, 0, 0,
         SEQ_ALL, 768, 3072, 1, 1, 1.f, 0);
  }

  // final layernorm -> f32 output
  ln_k<<<SEQ_ALL, 256, 0, stream>>>(xbuf, norm_g, norm_b, nullptr, (float*)d_out);
}
